// LinearAttention_8607114461865
// MI455X (gfx1250) — compile-verified, matrix operand order checked by reading, not run on hardware
//
#include <hip/hip_runtime.h>

// ---------------------------------------------------------------------------
// LinearAttention (collapsed form) for MI455X / gfx1250, wave32, bf16 WMMA.
// B=4 S=4096 H=1024 NH=16 HD=64
// ---------------------------------------------------------------------------

typedef __attribute__((ext_vector_type(16))) __bf16 v16bf;
typedef __attribute__((ext_vector_type(8)))  float  v8f;
typedef __attribute__((ext_vector_type(8)))  int    v8i;

#define B_   4
#define S_   4096
#define H_   1024
#define NH_  16
#define HD_  64
#define NTOK (B_*S_)          // 16384
#define XROW 1040             // LDS row stride in bf16 elems (pad to dodge bank conflicts)

__device__ __forceinline__ unsigned short f2bf(float f) {
  unsigned int u = __float_as_uint(f);
  unsigned int r = u + 0x7fffu + ((u >> 16) & 1u);   // round-to-nearest-even
  return (unsigned short)(r >> 16);
}

// ---- prep: f32 -> bf16 ----------------------------------------------------
__global__ void k_convert_bf16(const float* __restrict__ src,
                               unsigned short* __restrict__ dst, int n) {
  for (int i = blockIdx.x * blockDim.x + threadIdx.x; i < n;
       i += gridDim.x * blockDim.x)
    dst[i] = f2bf(src[i]);
}

// ---- prep: Wqsum[h][c] = sum_d Wq[h*64+d][c] (bf16); bqsum[h] -------------
__global__ void k_wqsum(const float* __restrict__ Wq, const float* __restrict__ bq,
                        unsigned short* __restrict__ Wqsb, float* __restrict__ bqs) {
  int i = blockIdx.x * blockDim.x + threadIdx.x;
  if (i < NH_ * H_) {
    int h = i >> 10, c = i & (H_ - 1);
    float s = 0.f;
    for (int d = 0; d < HD_; ++d) s += Wq[(size_t)(h * HD_ + d) * H_ + c];
    Wqsb[h * H_ + c] = f2bf(s);
  }
  if (i < NH_) {
    float s = 0.f;
    for (int d = 0; d < HD_; ++d) s += bq[i * HD_ + d];
    bqs[i] = s;
  }
}

// ---- prep: WoRowT[h][j] = sum_d Wo[j][h*64+d] -----------------------------
__global__ void k_worow(const float* __restrict__ Wo, float* __restrict__ WoRowT) {
  int i = blockIdx.x * blockDim.x + threadIdx.x;
  if (i >= NH_ * H_) return;
  int h = i >> 10, j = i & (H_ - 1);
  float s = 0.f;
  for (int d = 0; d < HD_; ++d) s += Wo[(size_t)j * H_ + h * HD_ + d];
  WoRowT[h * H_ + j] = s;
}

// ---------------------------------------------------------------------------
// Fused K/V projection + per-head dot + qsum.
// Block = 128 threads (4 waves). Block tile = 64 tokens staged in LDS via
// async global->LDS loads (ASYNCcnt). Wave w: qsum for token tile w, and
// dots for heads 4w..4w+3 over all 4 token tiles (4x B-operand reuse).
// WMMA C/D layout: VGPR r, lanes 0-15 -> (M=r, N=lane); lanes 16-31 -> (M=r+8).
// ---------------------------------------------------------------------------
__global__ __launch_bounds__(128)
void k_fused(const unsigned short* __restrict__ Xb,
             const unsigned short* __restrict__ Wkb,
             const unsigned short* __restrict__ Wvb,
             const unsigned short* __restrict__ Wqsb,
             const float* __restrict__ bk, const float* __restrict__ bv,
             const float* __restrict__ bqs, const float* __restrict__ mask,
             float* __restrict__ dots, float* __restrict__ qsum) {
  extern __shared__ unsigned short xs[];
  const int tid    = threadIdx.x;
  const int wave   = tid >> 5;
  const int lane   = tid & 31;
  const int nl     = lane & 15;     // N column (and A-matrix M row)
  const int half   = lane >> 4;
  const int token0 = blockIdx.x * 64;
  const int b      = token0 >> 12;  // /S_: 64-token tile never crosses batch

  // ---- stage 64x1024 bf16 X tile into LDS with async global->LDS b128 ----
  // Generic shared-pointer low 32 bits == wave-relative LDS byte offset.
  {
    const unsigned ldsbase = (unsigned)(unsigned long long)(void*)xs;
    for (int i = tid; i < 64 * 128; i += 128) {          // 16B chunks
      int row = i >> 7, ch = i & 127;
      unsigned la = ldsbase + (unsigned)(row * (XROW * 2) + ch * 16);
      unsigned long long ga =
          (unsigned long long)(Xb + (size_t)(token0 + row) * H_ + ch * 8);
      asm volatile("global_load_async_to_lds_b128 %0, %1, off"
                   :: "v"(la), "v"(ga) : "memory");
    }
    asm volatile("s_wait_asynccnt 0x0" ::: "memory");
  }
  __syncthreads();

  // ---------------- qsum for this wave's 16-token tile ----------------
  {
    const int arow = wave * 16 + nl;
    const float bqn = bqs[nl];
    v8f acc;
#pragma unroll
    for (int r = 0; r < 8; ++r) acc[r] = bqn;    // bias in C accumulator
    const unsigned short* bqrow = Wqsb + nl * H_ + half * 16;
#pragma unroll 4
    for (int ks = 0; ks < 32; ++ks) {
      v16bf A = *(const v16bf*)(&xs[arow * XROW + ks * 32 + half * 16]);
      v16bf Bq = *(const v16bf*)(bqrow + ks * 32);
      acc = __builtin_amdgcn_wmma_f32_16x16x32_bf16(false, A, false, Bq,
                                                    (short)0, acc, false, false);
    }
#pragma unroll
    for (int r = 0; r < 8; ++r) {
      int tok = token0 + wave * 16 + r + half * 8;
      int s = tok & (S_ - 1);
      qsum[((b * NH_ + nl) << 12) + s] = acc[r];
    }
  }

  // ------------- K/V projection + dot, 4 heads per wave ---------------
  for (int hh = 0; hh < 4; ++hh) {
    const int h = wave * 4 + hh;
    float pT[4][8];
#pragma unroll
    for (int tt = 0; tt < 4; ++tt)
#pragma unroll
      for (int r = 0; r < 8; ++r) pT[tt][r] = 0.f;

    for (int t = 0; t < 4; ++t) {                // 4 N-tiles of 16 = 64 head dims
      const int ncol = h * HD_ + t * 16 + nl;
      const float bkv = bk[ncol];
      const float bvv = bv[ncol];
      v8f aK[4], aV[4];
#pragma unroll
      for (int tt = 0; tt < 4; ++tt)
#pragma unroll
        for (int r = 0; r < 8; ++r) { aK[tt][r] = bkv; aV[tt][r] = bvv; }

      const unsigned short* bkrow = Wkb + (size_t)ncol * H_ + half * 16;
      const unsigned short* bvrow = Wvb + (size_t)ncol * H_ + half * 16;
      __builtin_prefetch(bkrow, 0, 0);           // global_prefetch: stream-in weights
      __builtin_prefetch(bvrow, 0, 0);
      for (int ks = 0; ks < 32; ++ks) {          // K dim: 1024 in steps of 32
        v16bf BK = *(const v16bf*)(bkrow + ks * 32);
        v16bf BV = *(const v16bf*)(bvrow + ks * 32);
#pragma unroll
        for (int tt = 0; tt < 4; ++tt) {         // 4 token tiles reuse B frags
          v16bf A = *(const v16bf*)(&xs[(tt * 16 + nl) * XROW + ks * 32 + half * 16]);
          aK[tt] = __builtin_amdgcn_wmma_f32_16x16x32_bf16(false, A, false, BK,
                                                           (short)0, aK[tt], false, false);
          aV[tt] = __builtin_amdgcn_wmma_f32_16x16x32_bf16(false, A, false, BV,
                                                           (short)0, aV[tt], false, false);
        }
      }
#pragma unroll
      for (int tt = 0; tt < 4; ++tt)
#pragma unroll
        for (int r = 0; r < 8; ++r) pT[tt][r] += aK[tt][r] * aV[tt][r];
    }

    // reduce over N (the 64 head dims live across the 16-lane halves)
#pragma unroll
    for (int tt = 0; tt < 4; ++tt)
#pragma unroll
      for (int r = 0; r < 8; ++r) {
        float v = pT[tt][r];
        v += __shfl_xor(v, 1, 32);
        v += __shfl_xor(v, 2, 32);
        v += __shfl_xor(v, 4, 32);
        v += __shfl_xor(v, 8, 32);
        pT[tt][r] = v;
      }
    if (nl == 0) {
#pragma unroll
      for (int tt = 0; tt < 4; ++tt)
#pragma unroll
        for (int r = 0; r < 8; ++r) {
          int tok = token0 + tt * 16 + r + half * 8;
          int s = tok & (S_ - 1);
          float mk = mask[b * S_ + s];
          dots[((b * NH_ + h) << 12) + s] = pT[tt][r] * mk * mk;
        }
    }
  }
}

// ---- cumsum over s per (b,h), fused with *qsum; also capture cum[-1] ------
__global__ __launch_bounds__(32)
void k_cumsum(const float* __restrict__ dots, const float* __restrict__ qsum,
              float* __restrict__ cvals, float* __restrict__ cumlast) {
  const int bh = blockIdx.x;
  const float* dr = dots + (size_t)bh * S_;
  const float* qr = qsum + (size_t)bh * S_;
  float* cr = cvals + (size_t)bh * S_;
  const int lane = threadIdx.x;
  float carry = 0.f;
  for (int base = 0; base < S_; base += 32) {
    float v = dr[base + lane];
    for (int off = 1; off < 32; off <<= 1) {     // Hillis-Steele inclusive scan
      float n = __shfl_up(v, off, 32);
      if (lane >= off) v += n;
    }
    float total = __shfl(v, 31, 32);
    v += carry;
    carry += total;
    cr[base + lane] = v * qr[base + lane];
  }
  if (lane == 31) cumlast[bh] = carry;           // cum[..., -1] (unscaled by qsum)
}

// ---- out = c(16) @ WoRowT(16x1024) + bo : memory-bound, 64MB write --------
__global__ void k_out(const float* __restrict__ cvals, const float* __restrict__ WoRowT,
                      const float* __restrict__ bo, float* __restrict__ out) {
  for (int idx = blockIdx.x * blockDim.x + threadIdx.x; idx < NTOK * H_;
       idx += gridDim.x * blockDim.x) {
    int j = idx & (H_ - 1);
    int tok = idx >> 10;
    int b = tok >> 12, s = tok & (S_ - 1);
    float acc = bo[j];
#pragma unroll
    for (int h = 0; h < NH_; ++h)
      acc = fmaf(cvals[((b * NH_ + h) << 12) + s], WoRowT[h * H_ + j], acc);
    out[idx] = acc;
  }
}

// ---- state = broadcast cum[b,h,-1] to (HD,HD) -----------------------------
__global__ void k_state(const float* __restrict__ cumlast, float* __restrict__ st) {
  for (int idx = blockIdx.x * blockDim.x + threadIdx.x; idx < B_ * NH_ * HD_ * HD_;
       idx += gridDim.x * blockDim.x)
    st[idx] = cumlast[idx >> 12];                // HD*HD = 4096
}

// ---------------------------------------------------------------------------
extern "C" void kernel_launch(void* const* d_in, const int* in_sizes, int n_in,
                              void* d_out, int out_size, void* d_ws, size_t ws_size,
                              hipStream_t stream) {
  const float* hidden = (const float*)d_in[0];
  const float* amask  = (const float*)d_in[1];
  const float* Wq     = (const float*)d_in[2];
  const float* bq     = (const float*)d_in[3];
  const float* Wk     = (const float*)d_in[4];
  const float* bk     = (const float*)d_in[5];
  const float* Wv     = (const float*)d_in[6];
  const float* bv     = (const float*)d_in[7];
  const float* Wo     = (const float*)d_in[8];
  const float* bo     = (const float*)d_in[9];
  float* out   = (float*)d_out;                  // (B,S,H)
  float* state = out + (size_t)NTOK * H_;        // (B,NH,HD,HD)

  char* ws = (char*)d_ws;
  size_t off = 0;
  auto alloc = [&](size_t bytes) -> char* {
    char* p = ws + off;
    off = (off + bytes + 255) & ~(size_t)255;
    return p;
  };
  unsigned short* Xb     = (unsigned short*)alloc((size_t)NTOK * H_ * 2);
  unsigned short* Wkb    = (unsigned short*)alloc((size_t)H_ * H_ * 2);
  unsigned short* Wvb    = (unsigned short*)alloc((size_t)H_ * H_ * 2);
  unsigned short* Wqsb   = (unsigned short*)alloc((size_t)NH_ * H_ * 2);
  float*          bqs    = (float*)alloc(NH_ * 4);
  float*          WoRowT = (float*)alloc((size_t)NH_ * H_ * 4);
  float*          dots   = (float*)alloc((size_t)B_ * NH_ * S_ * 4);
  float*          qsum   = (float*)alloc((size_t)B_ * NH_ * S_ * 4);
  float*          cvals  = (float*)alloc((size_t)B_ * NH_ * S_ * 4);
  float*          cumlast= (float*)alloc(B_ * NH_ * 4);

  k_convert_bf16<<<2048, 256, 0, stream>>>(hidden, Xb, NTOK * H_);
  k_convert_bf16<<<512, 256, 0, stream>>>(Wk, Wkb, H_ * H_);
  k_convert_bf16<<<512, 256, 0, stream>>>(Wv, Wvb, H_ * H_);
  k_wqsum<<<64, 256, 0, stream>>>(Wq, bq, Wqsb, bqs);
  k_worow<<<64, 256, 0, stream>>>(Wo, WoRowT);

  k_fused<<<NTOK / 64, 128, 64 * XROW * 2, stream>>>(
      Xb, Wkb, Wvb, Wqsb, bk, bv, bqs, amask, dots, qsum);

  k_cumsum<<<B_ * NH_, 32, 0, stream>>>(dots, qsum, cvals, cumlast);
  k_out<<<8192, 256, 0, stream>>>(cvals, WoRowT, bo, out);
  k_state<<<512, 256, 0, stream>>>(cumlast, state);
}
